// GCN_22213570854912
// MI455X (gfx1250) — matrix operand order repaired and checked by reading, not compiled, not run on hardware
//
#include <hip/hip_runtime.h>
#include <hip/hip_bf16.h>
#include <math.h>

// ---------------- problem constants ----------------
#define N_NODES 100000
#define N_EDGES 1600000
#define NFEAT   512
#define NHID    64
#define NCLASS  40
#define N_MT    6250              // N_NODES / 16 row-tiles

// ---------------- workspace layout (bytes) ----------------
#define OFF_FW1  0u                                  // 16*4*32*16 bf16 = 64 KB
#define OFF_FW2  65536u                              // 2*3*32*16 bf16 = 6 KB
#define OFF_T0   131072u                             // x@W1   [N,64] f32 = 25.6 MB (aliased by t2)
#define OFF_H    (131072u + 25600000u)               // spmm1  [N,64] f32 = 25.6 MB (aliased by o)

typedef __attribute__((ext_vector_type(16))) __bf16 v16bf;
typedef __attribute__((ext_vector_type(8)))  float  v8f;
typedef __attribute__((ext_vector_type(4)))  float  v4f;

// ---------------------------------------------------------------------------
// Async global->LDS stage of `bytes` (multiple of 2048) with 128 threads.
// Uses CDNA5 GLOBAL_LOAD_ASYNC_TO_LDS_B128 (ASYNCcnt) + s_wait_asynccnt.
__device__ __forceinline__ void stage_to_lds_128(const char* gsrc, void* lds,
                                                 int bytes, int tid) {
    unsigned lbase = (unsigned)(size_t)lds;          // low 32 bits = LDS offset
    for (int off = tid * 16; off < bytes; off += 128 * 16) {
        unsigned           loff  = lbase + off;
        unsigned long long gaddr = (unsigned long long)(gsrc + off);
        asm volatile("global_load_async_to_lds_b128 %0, %1, off"
                     :: "v"(loff), "v"(gaddr) : "memory");
    }
    asm volatile("s_wait_asynccnt 0x0" ::: "memory");
    __syncthreads();
}

// ---------------------------------------------------------------------------
// Pack W1 [512x64] f32 -> bf16 WMMA B-fragments.
// B (32x16 bf16 per tile): lane l holds column n=l&15, h=l>>4;
// element e (0..15) of the lane's v16bf is k_local = (e<8) ? 8h+e : 16+8h+(e-8).
// storage: frag[((kt*4+nt)*32 + l)*16 + e]
__global__ __launch_bounds__(256) void prep_w1(const float* __restrict__ W1,
                                               __bf16* __restrict__ frag) {
    int idx = blockIdx.x * 256 + threadIdx.x;      // 0..32767
    int e  = idx & 15;
    int l  = (idx >> 4) & 31;
    int nt = (idx >> 9) & 3;
    int kt = idx >> 11;
    int n     = l & 15;
    int khalf = (l >> 4) * 16;
    int k     = kt * 32 + (e & 8) * 2 + (khalf >> 1) + (e & 7);
    int col   = nt * 16 + n;
    frag[idx] = (__bf16)W1[k * NHID + col];
}

// Pack W2 [64x40] -> bf16 fragments, N padded to 48 (cols >= 40 are zero).
__global__ __launch_bounds__(256) void prep_w2(const float* __restrict__ W2,
                                               __bf16* __restrict__ frag) {
    int idx = blockIdx.x * 256 + threadIdx.x;      // 0..3071
    if (idx >= 2 * 3 * 32 * 16) return;
    int e  = idx & 15;
    int l  = (idx >> 4) & 31;
    int nt = (idx >> 9) % 3;
    int kt = idx / (3 * 512);
    int n     = l & 15;
    int khalf = (l >> 4) * 16;
    int k     = kt * 32 + (e & 8) * 2 + (khalf >> 1) + (e & 7);
    int col   = nt * 16 + n;
    frag[idx] = (col < NCLASS) ? (__bf16)W2[k * NCLASS + col] : (__bf16)0.0f;
}

// ---------------------------------------------------------------------------
// Build per-lane bf16 A-fragment from 4 f32 quads (16-bit A 16x32 layout).
__device__ __forceinline__ v16bf make_a(v4f a0, v4f a1, v4f a2, v4f a3) {
    v16bf a;
#pragma unroll
    for (int i = 0; i < 4; ++i) {
        a[i]      = (__bf16)a0[i];
        a[4 + i]  = (__bf16)a1[i];
        a[8 + i]  = (__bf16)a2[i];
        a[12 + i] = (__bf16)a3[i];
    }
    return a;
}

// ---------------------------------------------------------------------------
// GEMM1: t0[N,64] = x[N,512] @ W1.
// 128 threads = 4 waves; W1 fragments staged async into LDS once per block.
// Each wave computes 4 M-tiles x all 4 N-tiles -> 16 accumulators, so every
// B fragment pulled from LDS feeds 4 WMMAs.
__global__ __launch_bounds__(128) void gemm1_wmma(const float* __restrict__ x,
                                                  const __bf16* __restrict__ fragW1,
                                                  float* __restrict__ t0) {
    __shared__ __bf16 lfrag[16 * 4 * 32 * 16];       // 64 KB
    stage_to_lds_128((const char*)fragW1, lfrag, 65536, threadIdx.x);

    const int lane = threadIdx.x & 31;
    const int wave = threadIdx.x >> 5;
    const int r    = lane & 15;
    const int ks   = lane >> 4;
    const int mt0  = (blockIdx.x * 4 + wave) * 4;    // first of 4 M-tiles

    const float* xrow[4];
#pragma unroll
    for (int j = 0; j < 4; ++j) {
        int row = (mt0 + j) * 16 + r;
        if (row >= N_NODES) row = N_NODES - 1;       // clamp: keep EXEC all-1s
        xrow[j] = x + (size_t)row * NFEAT + ks * 8;
    }

    v8f acc[4][4] = {};
    for (int kt = 0; kt < 16; ++kt) {
        v16bf a[4];
#pragma unroll
        for (int j = 0; j < 4; ++j) {
            __builtin_prefetch(xrow[j] + (kt + 1) * 32, 0, 3);
            a[j] = make_a(*(const v4f*)(xrow[j] + kt * 32 + 0),
                          *(const v4f*)(xrow[j] + kt * 32 + 4),
                          *(const v4f*)(xrow[j] + kt * 32 + 16),
                          *(const v4f*)(xrow[j] + kt * 32 + 20));
        }
#pragma unroll
        for (int nt = 0; nt < 4; ++nt) {
            v16bf b = *(const v16bf*)(lfrag + ((kt * 4 + nt) * 32 + lane) * 16);
#pragma unroll
            for (int j = 0; j < 4; ++j)
                acc[j][nt] = __builtin_amdgcn_wmma_f32_16x16x32_bf16(
                    false, a[j], false, b, (short)0, acc[j][nt], false, false);
        }
    }
    // C layout: VGPR i, lanes0-15 -> (M=i, N=lane); lanes16-31 -> (M=8+i, N=lane-16)
#pragma unroll
    for (int j = 0; j < 4; ++j) {
        int mt = mt0 + j;
        if (mt >= N_MT) continue;                    // after all WMMAs: safe
#pragma unroll
        for (int i = 0; i < 8; ++i) {
            int row = i + 8 * ks;
            float* orow = t0 + (size_t)(mt * 16 + row) * NHID + r;
            orow[0]  = acc[j][0][i];
            orow[16] = acc[j][1][i];
            orow[32] = acc[j][2][i];
            orow[48] = acc[j][3][i];
        }
    }
}

// ---------------------------------------------------------------------------
// GEMM2: t2[N,40] = x1[N,64] @ W2 (bf16 fragments, N padded to 48).
__global__ __launch_bounds__(128) void gemm2_wmma(const float* __restrict__ x1,
                                                  const __bf16* __restrict__ fragW2,
                                                  float* __restrict__ t2) {
    __shared__ __bf16 lfrag[2 * 3 * 32 * 16];        // 6 KB
    stage_to_lds_128((const char*)fragW2, lfrag, 6144, threadIdx.x);

    const int lane = threadIdx.x & 31;
    const int wave = threadIdx.x >> 5;
    const int r    = lane & 15;
    const int ks   = lane >> 4;
    const int mt0  = (blockIdx.x * 4 + wave) * 4;

    const float* xrow[4];
#pragma unroll
    for (int j = 0; j < 4; ++j) {
        int row = (mt0 + j) * 16 + r;
        if (row >= N_NODES) row = N_NODES - 1;
        xrow[j] = x1 + (size_t)row * NHID + ks * 8;
    }

    v8f acc[4][3] = {};
#pragma unroll
    for (int kt = 0; kt < 2; ++kt) {
        v16bf a[4];
#pragma unroll
        for (int j = 0; j < 4; ++j)
            a[j] = make_a(*(const v4f*)(xrow[j] + kt * 32 + 0),
                          *(const v4f*)(xrow[j] + kt * 32 + 4),
                          *(const v4f*)(xrow[j] + kt * 32 + 16),
                          *(const v4f*)(xrow[j] + kt * 32 + 20));
#pragma unroll
        for (int nt = 0; nt < 3; ++nt) {
            v16bf b = *(const v16bf*)(lfrag + ((kt * 3 + nt) * 32 + lane) * 16);
#pragma unroll
            for (int j = 0; j < 4; ++j)
                acc[j][nt] = __builtin_amdgcn_wmma_f32_16x16x32_bf16(
                    false, a[j], false, b, (short)0, acc[j][nt], false, false);
        }
    }
#pragma unroll
    for (int j = 0; j < 4; ++j) {
        int mt = mt0 + j;
        if (mt >= N_MT) continue;
#pragma unroll
        for (int i = 0; i < 8; ++i) {
            int row = i + 8 * ks;
            float* orow = t2 + (size_t)(mt * 16 + row) * NCLASS;
            orow[r]      = acc[j][0][i];             // cols 0..15
            orow[16 + r] = acc[j][1][i];             // cols 16..31
            if (r < 8) orow[32 + r] = acc[j][2][i];  // cols 32..39 (pad dropped)
        }
    }
}

// ---------------------------------------------------------------------------
// SpMM d=64: wave per edge, lane covers 2 features via one b64 load.
__global__ __launch_bounds__(256) void spmm64(const float* __restrict__ t,
                                              const int* __restrict__ src,
                                              const int* __restrict__ dst,
                                              const float* __restrict__ ew,
                                              float* __restrict__ out) {
    int w    = (blockIdx.x * 256 + threadIdx.x) >> 5;   // edge id (grid exact)
    int lane = threadIdx.x & 31;
    int s = src[w];
    int d = dst[w];
    float wgt = ew[w];
    const float* ts = t + (size_t)s * NHID + lane * 2;
    float* od = out + (size_t)d * NHID + lane * 2;
    float2 v = *(const float2*)ts;
    unsafeAtomicAdd(od + 0, wgt * v.x);
    unsafeAtomicAdd(od + 1, wgt * v.y);
}

// SpMM d=40: wave per edge; lanes 0..31 -> f=lane, lanes 0..7 also f=32+lane.
__global__ __launch_bounds__(256) void spmm40(const float* __restrict__ t,
                                              const int* __restrict__ src,
                                              const int* __restrict__ dst,
                                              const float* __restrict__ ew,
                                              float* __restrict__ out) {
    int w    = (blockIdx.x * 256 + threadIdx.x) >> 5;
    int lane = threadIdx.x & 31;
    int s = src[w];
    int d = dst[w];
    float wgt = ew[w];
    const float* ts = t + (size_t)s * NCLASS;
    float* od = out + (size_t)d * NCLASS;
    unsafeAtomicAdd(od + lane, wgt * ts[lane]);
    if (lane < 8) unsafeAtomicAdd(od + 32 + lane, wgt * ts[32 + lane]);
}

// ---------------------------------------------------------------------------
// x1 = dropout(relu(h + b1)); written straight into d_out slot (gemm2 input).
__global__ __launch_bounds__(256) void relu_bias_drop(const float* __restrict__ h,
                                                      const float* __restrict__ b1,
                                                      const float* __restrict__ mask,
                                                      float* __restrict__ x1) {
    size_t i = (size_t)blockIdx.x * 256 + threadIdx.x;  // N*64 = 6.4M (grid exact)
    int f = (int)(i & 63);
    float v = h[i] + b1[f];
    v = v > 0.0f ? v : 0.0f;
    float keep = mask[i] > 0.5f ? 2.0f : 0.0f;          // 1/(1-0.5) = 2
    x1[i] = v * keep;
}

// out = log_softmax(o + b2) ; thread per node.
__global__ __launch_bounds__(256) void bias_logsoftmax(const float* __restrict__ o,
                                                       const float* __restrict__ b2,
                                                       float* __restrict__ out) {
    int n = blockIdx.x * 256 + threadIdx.x;
    if (n >= N_NODES) return;
    const float* row = o + (size_t)n * NCLASS;
    float vals[NCLASS];
    float m = -INFINITY;
#pragma unroll
    for (int c = 0; c < NCLASS; ++c) {
        vals[c] = row[c] + b2[c];
        m = fmaxf(m, vals[c]);
    }
    float s = 0.0f;
#pragma unroll
    for (int c = 0; c < NCLASS; ++c) s += __expf(vals[c] - m);
    float lse = m + __logf(s);
    float* orow = out + (size_t)n * NCLASS;
#pragma unroll
    for (int c = 0; c < NCLASS; ++c) orow[c] = vals[c] - lse;
}

// ---------------------------------------------------------------------------
extern "C" void kernel_launch(void* const* d_in, const int* in_sizes, int n_in,
                              void* d_out, int out_size, void* d_ws, size_t ws_size,
                              hipStream_t stream) {
    const float* x    = (const float*)d_in[0];
    const int*   src  = (const int*)  d_in[1];
    const int*   dst  = (const int*)  d_in[2];
    const float* ew   = (const float*)d_in[3];
    const float* W1   = (const float*)d_in[4];
    const float* b1   = (const float*)d_in[5];
    const float* W2   = (const float*)d_in[6];
    const float* b2   = (const float*)d_in[7];
    const float* mask = (const float*)d_in[8];

    char* ws = (char*)d_ws;
    __bf16* fragW1 = (__bf16*)(ws + OFF_FW1);
    __bf16* fragW2 = (__bf16*)(ws + OFF_FW2);
    float*  t0     = (float*)(ws + OFF_T0);   // x@W1
    float*  h      = (float*)(ws + OFF_H);    // spmm1 accumulator
    float*  t2     = (float*)(ws + OFF_T0);   // x1@W2 (aliases t0: dead by then)
    float*  oacc   = (float*)(ws + OFF_H);    // spmm2 accumulator (aliases h)

    float* logits = (float*)d_out;                             // [N, 40]
    float* x1     = (float*)d_out + (size_t)N_NODES * NCLASS;  // [N, 64]

    // weight fragment packing
    prep_w1<<<128, 256, 0, stream>>>(W1, fragW1);
    prep_w2<<<12, 256, 0, stream>>>(W2, fragW2);

    const int gemm_blocks = (N_MT + 15) / 16;   // 391: 4 waves x 4 M-tiles each

    // layer 1
    gemm1_wmma<<<gemm_blocks, 128, 0, stream>>>(x, fragW1, t0);
    hipMemsetAsync(h, 0, (size_t)N_NODES * NHID * sizeof(float), stream);
    spmm64<<<N_EDGES / 8, 256, 0, stream>>>(t0, src, dst, ew, h);
    relu_bias_drop<<<(N_NODES * NHID) / 256, 256, 0, stream>>>(h, b1, mask, x1);

    // layer 2
    gemm2_wmma<<<gemm_blocks, 128, 0, stream>>>(x1, fragW2, t2);
    hipMemsetAsync(oacc, 0, (size_t)N_NODES * NCLASS * sizeof(float), stream);
    spmm40<<<N_EDGES / 8, 256, 0, stream>>>(t2, src, dst, ew, oacc);
    bias_logsoftmax<<<(N_NODES + 255) / 256, 256, 0, stream>>>(oacc, b2, logits);
}
